// SwitchMLP_89687507076290
// MI455X (gfx1250) — matrix operand order repaired and checked by reading, not compiled, not run on hardware
//
#include <hip/hip_runtime.h>

// SwitchMLP top-1 MoE for MI455X (gfx1250), wave32 + WMMA bf16.
// Pipeline: convert x->bf16, transpose-convert weights->bf16 [N][K],
// router -> per-expert token lists -> fused SwiGLU GEMM -> W3 GEMM.
// ws: counts|offsets|eidx|pos|prob|list|xb|w1t|w2t|w3t|h1  (~99 MB)

#define T_TOK 4096
#define DIM   768
#define HID   2048
#define NE    8

typedef __attribute__((ext_vector_type(16))) __bf16 v16bf;
typedef __attribute__((ext_vector_type(8)))  float  v8f;

union BFrag { v16bf v; unsigned short s[16]; uint4 q[2]; };

static __device__ inline unsigned short f32_to_bf16(float f) {
  unsigned int u = __float_as_uint(f);
  u += 0x7FFFu + ((u >> 16) & 1u);     // round-to-nearest-even
  return (unsigned short)(u >> 16);
}
static __device__ inline float bf16_to_f32(unsigned short h) {
  return __uint_as_float(((unsigned int)h) << 16);
}
static __device__ inline unsigned int pack2(float lo, float hi) {
  return (unsigned int)f32_to_bf16(lo) | ((unsigned int)f32_to_bf16(hi) << 16);
}

// ---------------- prep: x f32 -> bf16 ----------------
__global__ void k_convx(const float* __restrict__ x, unsigned short* __restrict__ xb) {
  size_t i = ((size_t)blockIdx.x * blockDim.x + threadIdx.x) * 8;
  float4 f0 = *(const float4*)(x + i);
  float4 f1 = *(const float4*)(x + i + 4);
  uint4 o;
  o.x = pack2(f0.x, f0.y); o.y = pack2(f0.z, f0.w);
  o.z = pack2(f1.x, f1.y); o.w = pack2(f1.z, f1.w);
  *(uint4*)(xb + i) = o;
}

// ---------------- prep: weight f32 [K][N] -> bf16 [N][K] (batched) ----------------
__global__ void k_tconv(const float* __restrict__ src, unsigned short* __restrict__ dst,
                        int K, int N) {
  const int b = blockIdx.z;
  const float* s = src + (size_t)b * K * N;
  unsigned short* d = dst + (size_t)b * N * K;
  __shared__ float tile[32][33];
  const int kBase = blockIdx.y * 32, nBase = blockIdx.x * 32;
  const int tr = threadIdx.x >> 3;        // 0..31
  const int tc = (threadIdx.x & 7) * 4;   // 0,4,..,28
  float4 v = *(const float4*)(s + (size_t)(kBase + tr) * N + nBase + tc);
  tile[tr][tc + 0] = v.x; tile[tr][tc + 1] = v.y;
  tile[tr][tc + 2] = v.z; tile[tr][tc + 3] = v.w;
  __syncthreads();
  // dst[nBase+tr][kBase+tc .. +3] = tile[tc..tc+3][tr]
  uint2 o = make_uint2(pack2(tile[tc + 0][tr], tile[tc + 1][tr]),
                       pack2(tile[tc + 2][tr], tile[tc + 3][tr]));
  *(uint2*)(d + (size_t)(nBase + tr) * K + kBase + tc) = o;
}

// ---------------- router ----------------
__global__ void k_init(int* counts) {
  if (threadIdx.x < NE) counts[threadIdx.x] = 0;
}

__global__ void k_router(const float* __restrict__ x, const float* __restrict__ rw,
                         const float* __restrict__ rb, int* __restrict__ eidx,
                         float* __restrict__ prob, int* __restrict__ pos,
                         int* __restrict__ counts) {
  const int wid  = threadIdx.x >> 5;
  const int lane = threadIdx.x & 31;
  const int t = blockIdx.x * 8 + wid;          // one wave32 per token
  if (t >= T_TOK) return;
  float acc[NE];
#pragma unroll
  for (int e = 0; e < NE; ++e) acc[e] = 0.f;
  const float* xr = x + (size_t)t * DIM;
  for (int d = lane; d < DIM; d += 32) {
    float xv = xr[d];
    const float* wr = rw + (size_t)d * NE;
#pragma unroll
    for (int e = 0; e < NE; ++e) acc[e] += xv * wr[e];
  }
#pragma unroll
  for (int e = 0; e < NE; ++e) {
#pragma unroll
    for (int off = 16; off > 0; off >>= 1)
      acc[e] += __shfl_xor(acc[e], off, 32);
  }
  if (lane == 0) {
    float mx = -1e30f; int mi = 0;
#pragma unroll
    for (int e = 0; e < NE; ++e) {
      acc[e] += rb[e];
      if (acc[e] > mx) { mx = acc[e]; mi = e; }
    }
    float s = 0.f;
#pragma unroll
    for (int e = 0; e < NE; ++e) s += __expf(acc[e] - mx);
    eidx[t] = mi;
    prob[t] = 1.0f / s;                        // max softmax prob
    pos[t]  = atomicAdd(&counts[mi], 1);
  }
}

__global__ void k_scan(const int* __restrict__ counts, int* __restrict__ offsets) {
  if (threadIdx.x == 0) {
    int o = 0;
    for (int e = 0; e < NE; ++e) { offsets[e] = o; o += counts[e]; }
  }
}

__global__ void k_scatter(const int* __restrict__ eidx, const int* __restrict__ pos,
                          const int* __restrict__ offsets, int* __restrict__ list) {
  int t = blockIdx.x * blockDim.x + threadIdx.x;
  if (t < T_TOK) list[offsets[eidx[t]] + pos[t]] = t;
}

// ---------------- WMMA fragment helpers (all-bf16, LDS-resident) ----------------
// A 16x32: lanes 0-15 row M=lane&15, K {0..7,16..23}; lanes 16-31 K {8..15,24..31}.
static __device__ inline v16bf frag_a(const unsigned short* xs, int rowBase, int lane) {
  const unsigned short* p = xs + (rowBase + (lane & 15)) * 32 + ((lane & 16) ? 8 : 0);
  BFrag f;
  f.q[0] = *(const uint4*)(p);
  f.q[1] = *(const uint4*)(p + 16);
  return f.v;
}
// B 32x16: lane = col n (mod 16); lanes 0-15 K {0..7,16..23}, lanes 16-31 K {8..15,24..31}.
// bs is [64 n][32 k] bf16 -> per-lane 2 x b128 (k runs of 8).
static __device__ inline v16bf frag_b(const unsigned short* bs, int nRowBase, int lane) {
  const unsigned short* p = bs + (nRowBase + (lane & 15)) * 32 + ((lane & 16) ? 8 : 0);
  BFrag f;
  f.q[0] = *(const uint4*)(p);
  f.q[1] = *(const uint4*)(p + 16);
  return f.v;
}

// ---------------- stage 1: H1 = silu(Xe@W1e) * (Xe@W2e) ----------------
// Block tile M=64, N=64; 8 waves: wm=wid&3 (16 rows), wn=wid>>2 (32 cols).
__global__ void k_stage1(const unsigned short* __restrict__ xb,
                         const unsigned short* __restrict__ w1t,
                         const unsigned short* __restrict__ w2t,
                         const int* __restrict__ list,
                         const int* __restrict__ counts,
                         const int* __restrict__ offsets,
                         unsigned short* __restrict__ h1) {
  const int e = blockIdx.z;
  const int Ne = counts[e];
  const int tileBase = blockIdx.y * 64;
  if (tileBase >= Ne) return;
  const int expOff = offsets[e];
  const int nBase = blockIdx.x * 64;
  const unsigned short* w1e = w1t + (size_t)e * HID * DIM;   // [HID n][DIM k]
  const unsigned short* w2e = w2t + (size_t)e * HID * DIM;

  __shared__ __align__(16) unsigned short xs[64 * 32];
  __shared__ __align__(16) unsigned short b1s[64 * 32];
  __shared__ __align__(16) unsigned short b2s[64 * 32];
  __shared__ int toks[64];

  const int tid  = threadIdx.x;
  const int lane = tid & 31;
  const int wid  = tid >> 5;
  const int wm   = wid & 3;            // 4 wave rows * 16
  const int wn   = wid >> 2;           // 2 wave cols * 32

  if (tid < 64) {
    int p = tileBase + tid;
    toks[tid] = (p < Ne) ? list[expOff + p] : -1;
  }
  __syncthreads();

  v8f acc1[2] = {}, acc2[2] = {};
  const int ldRow = tid >> 2;          // 0..63
  const int ldK   = (tid & 3) * 8;     // 0,8,16,24
  const int tok   = toks[ldRow];
  const unsigned short* xsrc = (tok >= 0) ? xb + (size_t)tok * DIM + ldK : nullptr;
  const unsigned short* b1src = w1e + (size_t)(nBase + ldRow) * DIM + ldK;
  const unsigned short* b2src = w2e + (size_t)(nBase + ldRow) * DIM + ldK;

  for (int k0 = 0; k0 < DIM; k0 += 32) {
    // stage A (gathered bf16 tokens) + both B tiles into LDS, 128-bit coalesced
    *(uint4*)(xs + ldRow * 32 + ldK) =
        xsrc ? *(const uint4*)(xsrc + k0) : make_uint4(0, 0, 0, 0);
    *(uint4*)(b1s + ldRow * 32 + ldK) = *(const uint4*)(b1src + k0);
    *(uint4*)(b2s + ldRow * 32 + ldK) = *(const uint4*)(b2src + k0);
    if (k0 + 32 < DIM) {
      __builtin_prefetch(b1src + k0 + 32, 0, 1);
      __builtin_prefetch(b2src + k0 + 32, 0, 1);
    }
    __syncthreads();

    v16bf a = frag_a(xs, wm * 16, lane);
#pragma unroll
    for (int s = 0; s < 2; ++s) {
      v16bf b1 = frag_b(b1s, wn * 32 + s * 16, lane);
      v16bf b2 = frag_b(b2s, wn * 32 + s * 16, lane);
      acc1[s] = __builtin_amdgcn_wmma_f32_16x16x32_bf16(false, a, false, b1,
                                                        (short)0, acc1[s], false, false);
      acc2[s] = __builtin_amdgcn_wmma_f32_16x16x32_bf16(false, a, false, b2,
                                                        (short)0, acc2[s], false, false);
    }
    __syncthreads();
  }

  const int hi8 = (lane & 16) ? 8 : 0;
#pragma unroll
  for (int s = 0; s < 2; ++s) {
    const int ncol = nBase + wn * 32 + s * 16 + (lane & 15);
#pragma unroll
    for (int r = 0; r < 8; ++r) {
      int p = tileBase + wm * 16 + r + hi8;
      if (p < Ne) {
        float g  = acc1[s][r];
        float sl = g / (1.0f + __expf(-g));            // silu
        h1[(size_t)(expOff + p) * HID + ncol] = f32_to_bf16(sl * acc2[s][r]);
      }
    }
  }
}

// ---------------- stage 2: out = bf16(H1 @ W3e) * prob ----------------
__global__ void k_stage2(const unsigned short* __restrict__ h1,
                         const unsigned short* __restrict__ w3t,
                         const int* __restrict__ list,
                         const int* __restrict__ counts,
                         const int* __restrict__ offsets,
                         const float* __restrict__ prob,
                         float* __restrict__ out) {
  const int e = blockIdx.z;
  const int Ne = counts[e];
  const int tileBase = blockIdx.y * 64;
  if (tileBase >= Ne) return;
  const int expOff = offsets[e];
  const int nBase = blockIdx.x * 64;
  const unsigned short* w3e = w3t + (size_t)e * DIM * HID;   // [DIM n][HID k]

  __shared__ __align__(16) unsigned short hs[64 * 32];
  __shared__ __align__(16) unsigned short bs[64 * 32];
  __shared__ int toks[64];
  __shared__ float pr[64];

  const int tid  = threadIdx.x;
  const int lane = tid & 31;
  const int wid  = tid >> 5;
  const int wm   = wid & 3;
  const int wn   = wid >> 2;

  if (tid < 64) {
    int p = tileBase + tid;
    int tk = (p < Ne) ? list[expOff + p] : -1;
    toks[tid] = tk;
    pr[tid] = (tk >= 0) ? prob[tk] : 0.f;
  }
  __syncthreads();

  v8f acc[2] = {};
  const int ldRow = tid >> 2;
  const int ldK   = (tid & 3) * 8;
  const bool rowOk = (tileBase + ldRow) < Ne;
  const unsigned short* asrc = h1 + (size_t)(expOff + tileBase + ldRow) * HID + ldK;
  const unsigned short* bsrc = w3e + (size_t)(nBase + ldRow) * HID + ldK;

  for (int k0 = 0; k0 < HID; k0 += 32) {
    *(uint4*)(hs + ldRow * 32 + ldK) =
        rowOk ? *(const uint4*)(asrc + k0) : make_uint4(0, 0, 0, 0);
    *(uint4*)(bs + ldRow * 32 + ldK) = *(const uint4*)(bsrc + k0);
    if (k0 + 32 < HID) __builtin_prefetch(bsrc + k0 + 32, 0, 1);
    __syncthreads();

    v16bf a = frag_a(hs, wm * 16, lane);
#pragma unroll
    for (int s = 0; s < 2; ++s) {
      v16bf b = frag_b(bs, wn * 32 + s * 16, lane);
      acc[s] = __builtin_amdgcn_wmma_f32_16x16x32_bf16(false, a, false, b,
                                                       (short)0, acc[s], false, false);
    }
    __syncthreads();
  }

  const int hi8 = (lane & 16) ? 8 : 0;
#pragma unroll
  for (int s = 0; s < 2; ++s) {
    const int ncol = nBase + wn * 32 + s * 16 + (lane & 15);
#pragma unroll
    for (int r = 0; r < 8; ++r) {
      int rowLocal = wm * 16 + r + hi8;
      int tk = toks[rowLocal];
      if (tk >= 0) {
        // reference: bf16-round the expert output, then scale by f32 prob
        out[(size_t)tk * DIM + ncol] = bf16_to_f32(f32_to_bf16(acc[s][r])) * pr[rowLocal];
      }
    }
  }
}

// ---------------- host launcher ----------------
extern "C" void kernel_launch(void* const* d_in, const int* in_sizes, int n_in,
                              void* d_out, int out_size, void* d_ws, size_t ws_size,
                              hipStream_t stream) {
  const float* x  = (const float*)d_in[0];   // [S,B,DIM] f32
  const float* rw = (const float*)d_in[1];   // [DIM,NE]
  const float* rb = (const float*)d_in[2];   // [NE]
  const float* w1 = (const float*)d_in[3];   // [NE,DIM,HID]
  const float* w2 = (const float*)d_in[4];   // [NE,DIM,HID]
  const float* w3 = (const float*)d_in[5];   // [NE,HID,DIM]
  float* out = (float*)d_out;

  char* ws = (char*)d_ws;
  int*   counts  = (int*)(ws);
  int*   offsets = (int*)(ws + 32);
  int*   eidx    = (int*)(ws + 64);
  int*   pos     = (int*)(ws + 64 + 4 * T_TOK);
  float* prob    = (float*)(ws + 64 + 8 * T_TOK);
  int*   list    = (int*)(ws + 64 + 12 * T_TOK);
  size_t off = (size_t)((64 + 16 * T_TOK + 255) & ~255);
  unsigned short* xb  = (unsigned short*)(ws + off); off += (size_t)T_TOK * DIM * 2;
  unsigned short* w1t = (unsigned short*)(ws + off); off += (size_t)NE * DIM * HID * 2;
  unsigned short* w2t = (unsigned short*)(ws + off); off += (size_t)NE * DIM * HID * 2;
  unsigned short* w3t = (unsigned short*)(ws + off); off += (size_t)NE * HID * DIM * 2;
  unsigned short* h1  = (unsigned short*)(ws + off);   // [T_TOK][HID] bf16

  // prep: bf16 conversion + weight transposition (bandwidth-bound, once per call)
  k_convx<<<(T_TOK * DIM) / (256 * 8), 256, 0, stream>>>(x, xb);
  {
    dim3 g1(HID / 32, DIM / 32, NE);                 // w1,w2: [DIM k][HID n] -> [HID][DIM]
    k_tconv<<<g1, 256, 0, stream>>>(w1, w1t, DIM, HID);
    k_tconv<<<g1, 256, 0, stream>>>(w2, w2t, DIM, HID);
    dim3 g3(DIM / 32, HID / 32, NE);                 // w3: [HID k][DIM n] -> [DIM][HID]
    k_tconv<<<g3, 256, 0, stream>>>(w3, w3t, HID, DIM);
  }

  k_init<<<1, 32, 0, stream>>>(counts);
  k_router<<<T_TOK / 8, 256, 0, stream>>>(x, rw, rb, eidx, prob, pos, counts);
  k_scan<<<1, 32, 0, stream>>>(counts, offsets);
  k_scatter<<<T_TOK / 256, 256, 0, stream>>>(eidx, pos, offsets, list);

  dim3 g1(HID / 64, T_TOK / 64, NE);
  k_stage1<<<g1, 256, 0, stream>>>(xb, w1t, w2t, list, counts, offsets, h1);

  dim3 g2(DIM / 64, T_TOK / 64, NE);
  k_stage2<<<g2, 256, 0, stream>>>(h1, w3t, list, counts, offsets, prob, out);
}
